// ProteinStructureEncoder_50714973831806
// MI455X (gfx1250) — matrix operand order repaired and checked by reading, not compiled, not run on hardware
//
#include <hip/hip_runtime.h>
#include <math.h>

// ---------------- problem constants (match reference) ----------------
constexpr int NN   = 20000;    // nodes
constexpr int EE   = 320000;   // edges
constexpr int NDIM = 64;       // node feature dim
constexpr int EDIM = 16;       // edge feature dim
constexpr int HID  = 256;
constexpr int HEADS = 8;
constexpr int CC   = 32;       // channels per head
constexpr int NB   = 16;       // graphs
constexpr int NL   = 3;        // layers
constexpr int FF   = EE + NN;  // edges + self loops

typedef __attribute__((ext_vector_type(16))) __bf16 v16bf;
typedef __attribute__((ext_vector_type(8)))  __bf16 v8bf;
typedef __attribute__((ext_vector_type(8)))  float  v8f;

// ---------------- WMMA bf16 GEMM: C[M,Nc] = A[M,K] @ W[K,Nc] + bias ----------------
// Requires K % 32 == 0 (callers zero-pad). One wave computes a 16x16 output tile.
// Fragment layouts per CDNA5 ISA 7.12.2 (wave32):
//  A 16x32: lane group kg=lane>>4 holds row M=(lane&15); elements 0..7 = K[kg*8 .. kg*8+7],
//           elements 8..15 = K[16+kg*8 .. 16+kg*8+7]  -> two contiguous 16B loads per lane
//  B 32x16: lane holds column N=(lane&15); elements 0..7 = K[kg*8 .. kg*8+7], 8..15 = K[16+kg*8 ..] (strided gather)
//  C/D 16x16: N=(lane&15), M = vgpr + 8*kg
__global__ __launch_bounds__(128) void gemm_bf16_wmma(
    const __bf16* __restrict__ A, const __bf16* __restrict__ W,
    const float* __restrict__ bias, float* __restrict__ C,
    int M, int K, int Nc)
{
    const int wave  = threadIdx.x >> 5;
    const int lane  = threadIdx.x & 31;
    const int mtile = blockIdx.x;
    const int ntile = blockIdx.y * 4 + wave;
    if (ntile * 16 >= Nc) return;

    const int kg = lane >> 4;            // lane group 0/1
    const int m  = mtile * 16 + (lane & 15);
    const int n0 = ntile * 16;

    const __bf16* arow = A + (long)m * K + kg * 8;
    const __bf16* bcol = W + (long)(kg * 8) * Nc + n0 + (lane & 15);

    v8f acc = {};
    for (int k0 = 0; k0 < K; k0 += 32) {
        v8bf alo = *(const v8bf*)(arow + k0);
        v8bf ahi = *(const v8bf*)(arow + k0 + 16);
        v16bf a  = __builtin_shufflevector(alo, ahi,
                       0, 1, 2, 3, 4, 5, 6, 7, 8, 9, 10, 11, 12, 13, 14, 15);
        v16bf b;
        #pragma unroll
        for (int s = 0; s < 8; ++s) {
            b[s]     = bcol[(long)(k0 + s) * Nc];
            b[s + 8] = bcol[(long)(k0 + 16 + s) * Nc];
        }

        acc = __builtin_amdgcn_wmma_f32_16x16x32_bf16(
                  false, a, false, b, (short)0, acc, false, false);
    }
    #pragma unroll
    for (int r = 0; r < 8; ++r) {
        int Mr   = mtile * 16 + r + 8 * kg;
        int Ncol = n0 + (lane & 15);
        C[(long)Mr * Nc + Ncol] = acc[r] + bias[Ncol];
    }
}

// ---------------- small utility kernels ----------------
__global__ void cvt_bf16(const float* __restrict__ in, __bf16* __restrict__ out, long n) {
    long i = (long)blockIdx.x * blockDim.x + threadIdx.x;
    if (i < n) out[i] = (__bf16)in[i];
}

// copy [nrows,16] f32 -> [nrows,32] bf16 (cols 16..31 pre-zeroed by memset)
__global__ void cvt_bf16_pad16to32(const float* __restrict__ in, __bf16* __restrict__ out,
                                   long nrows) {
    long t = (long)blockIdx.x * blockDim.x + threadIdx.x;
    if (t >= nrows * 16) return;
    long e = t >> 4;
    int  k = (int)(t & 15);
    out[e * 32 + k] = (__bf16)in[t];
}

__global__ void fill_f32(float* __restrict__ p, float v, long n) {
    long i = (long)blockIdx.x * blockDim.x + threadIdx.x;
    if (i < n) p[i] = v;
}

// deg[d] += 1 ; loop_ea[d,k] += ea[e,k]   (computed once; invariant across layers)
__global__ void deg_loop_ea(const int* __restrict__ dst, const float* __restrict__ ea,
                            float* __restrict__ deg, float* __restrict__ loop_ea) {
    long t = (long)blockIdx.x * blockDim.x + threadIdx.x;
    if (t >= (long)EE * EDIM) return;
    int e = (int)(t >> 4), k = (int)(t & 15);
    int d = dst[e];
    atomicAdd(&loop_ea[(long)d * EDIM + k], ea[t]);
    if (k == 0) atomicAdd(&deg[d], 1.0f);
}

__global__ void loop_ea_norm(float* __restrict__ loop_ea, const float* __restrict__ deg) {
    int t = blockIdx.x * blockDim.x + threadIdx.x;
    if (t >= NN * EDIM) return;
    loop_ea[t] /= fmaxf(deg[t >> 4], 1.0f);
}

// ---------------- attention logits (fused edge-embedding GEMM) ----------------
// thread = (edge_or_loop f, head). eemb computed on the fly from ea[f,0:16] @ We (LDS).
// Also performs segment-max into mmax via the int/uint float-atomic-max trick.
__global__ __launch_bounds__(256) void attn_logits(
    const float* __restrict__ xl, const float* __restrict__ xr,
    const float* __restrict__ ea, const float* __restrict__ loop_ea,
    const int* __restrict__ src, const int* __restrict__ dst,
    const float* __restrict__ We,   // [16,256]
    const float* __restrict__ att,  // [8,32]
    float* __restrict__ logits, float* __restrict__ mmax)
{
    __shared__ float We_s[EDIM * HID];
    __shared__ float att_s[HEADS * CC];
    for (int i = threadIdx.x; i < EDIM * HID; i += 256) We_s[i] = We[i];
    if (threadIdx.x < HEADS * CC) att_s[threadIdx.x] = att[threadIdx.x];
    __syncthreads();

    long gid = (long)blockIdx.x * 256 + threadIdx.x;
    int f  = (int)(gid >> 3);
    int hd = (int)(gid & 7);
    if (f >= FF) return;

    int sf, df; const float* eat;
    if (f < EE) { sf = src[f]; df = dst[f]; eat = ea + (long)f * EDIM; }
    else        { sf = f - EE; df = sf;     eat = loop_ea + (long)sf * EDIM; }

    float ev[EDIM];
    #pragma unroll
    for (int k = 0; k < EDIM; ++k) ev[k] = eat[k];

    const float4* xlp = (const float4*)(xl + (long)sf * HID + hd * CC);
    const float4* xrp = (const float4*)(xr + (long)df * HID + hd * CC);

    float acc = 0.0f;
    #pragma unroll
    for (int cq = 0; cq < CC / 4; ++cq) {
        float4 vl = xlp[cq];
        float4 vr = xrp[cq];
        const float xls[4] = {vl.x, vl.y, vl.z, vl.w};
        const float xrs[4] = {vr.x, vr.y, vr.z, vr.w};
        #pragma unroll
        for (int cp = 0; cp < 4; ++cp) {
            int c = cq * 4 + cp;
            float em = 0.0f;
            #pragma unroll
            for (int k = 0; k < EDIM; ++k)
                em = fmaf(ev[k], We_s[k * HID + hd * CC + c], em);
            float z = xls[cp] + xrs[cp] + em;
            z = (z > 0.0f) ? z : 0.2f * z;             // leaky_relu(0.2)
            acc = fmaf(att_s[hd * CC + c], z, acc);
        }
    }
    logits[(long)f * HEADS + hd] = acc;

    // segment max over incoming edges of df (init = -inf)
    float* addr = &mmax[(long)df * HEADS + hd];
    if (acc >= 0.0f) atomicMax((int*)addr, __float_as_int(acc));
    else             atomicMin((unsigned int*)addr, __float_as_uint(acc));
}

// denom[d,h] += exp(logit - max)
__global__ void seg_denom(const int* __restrict__ dst, const float* __restrict__ logits,
                          const float* __restrict__ mmax, float* __restrict__ denom) {
    long t = (long)blockIdx.x * blockDim.x + threadIdx.x;
    if (t >= (long)FF * HEADS) return;
    int f = (int)(t >> 3), hd = (int)(t & 7);
    int df = (f < EE) ? dst[f] : (f - EE);
    float ex = expf(logits[t] - mmax[(long)df * HEADS + hd]);
    atomicAdd(&denom[(long)df * HEADS + hd], ex);
}

// hnew = h + conv_bias (residual + bias, then edges accumulate into it)
__global__ void init_hnew(const float* __restrict__ h, const float* __restrict__ cbias,
                          float* __restrict__ hnew) {
    long i = (long)blockIdx.x * blockDim.x + threadIdx.x;
    if (i < (long)NN * HID) hnew[i] = h[i] + cbias[i & (HID - 1)];
}

// hnew[df] += alpha(f,h) * xl[sf]  — thread = (f, channel-in-head), loop over heads
__global__ __launch_bounds__(256) void aggregate(
    const float* __restrict__ xl, const float* __restrict__ logits,
    const float* __restrict__ mmax, const float* __restrict__ denom,
    const int* __restrict__ src, const int* __restrict__ dst,
    float* __restrict__ hnew)
{
    long gid = (long)blockIdx.x * 256 + threadIdx.x;
    int f = (int)(gid >> 5);
    int c = (int)(gid & 31);
    if (f >= FF) return;
    int sf, df;
    if (f < EE) { sf = src[f]; df = dst[f]; } else { sf = f - EE; df = sf; }
    #pragma unroll
    for (int hd = 0; hd < HEADS; ++hd) {
        float alpha = expf(logits[(long)f * HEADS + hd] - mmax[(long)df * HEADS + hd])
                      / denom[(long)df * HEADS + hd];
        atomicAdd(&hnew[(long)df * HID + hd * CC + c],
                  alpha * xl[(long)sf * HID + hd * CC + c]);
    }
}

// ---------------- pooling + MLP helpers ----------------
__global__ void pool_sum(const float* __restrict__ h, const int* __restrict__ batch,
                         float* __restrict__ hgsum, float* __restrict__ cnt) {
    long t = (long)blockIdx.x * blockDim.x + threadIdx.x;
    if (t >= (long)NN * HID) return;
    int n = (int)(t >> 8), c = (int)(t & 255);
    int b = batch[n];
    atomicAdd(&hgsum[b * HID + c], h[t]);
    if (c == 0) atomicAdd(&cnt[b], 1.0f);
}

__global__ void pool_norm(const float* __restrict__ hgsum, const float* __restrict__ cnt,
                          float* __restrict__ hg) {
    int i = blockIdx.x * blockDim.x + threadIdx.x;
    if (i >= NB * HID) return;
    hg[i] = hgsum[i] / fmaxf(cnt[i >> 8], 1.0f);
}

__global__ void gelu_exact(float* __restrict__ p, long n) {
    long i = (long)blockIdx.x * blockDim.x + threadIdx.x;
    if (i >= n) return;
    float v = p[i];
    p[i] = 0.5f * v * (1.0f + erff(v * 0.70710678118654752f));
}

// ---------------- host orchestration ----------------
static inline long cdiv(long a, long b) { return (a + b - 1) / b; }

extern "C" void kernel_launch(void* const* d_in, const int* in_sizes, int n_in,
                              void* d_out, int out_size, void* d_ws, size_t ws_size,
                              hipStream_t stream)
{
    const float* x      = (const float*)d_in[0];
    const int*   eidx   = (const int*)d_in[1];
    const float* eattr  = (const float*)d_in[2];
    const int*   batch  = (const int*)d_in[3];
    const float* node_W = (const float*)d_in[4];
    const float* node_b = (const float*)d_in[5];
    const float* edge_W = (const float*)d_in[6];
    const float* edge_b = (const float*)d_in[7];
    const float* Wl     = (const float*)d_in[8];
    const float* bl     = (const float*)d_in[9];
    const float* Wr     = (const float*)d_in[10];
    const float* br     = (const float*)d_in[11];
    const float* WeAll  = (const float*)d_in[12];
    const float* attAll = (const float*)d_in[13];
    const float* cbias  = (const float*)d_in[14];
    const float* W1     = (const float*)d_in[15];
    const float* b1     = (const float*)d_in[16];
    const float* W2     = (const float*)d_in[17];
    const float* b2     = (const float*)d_in[18];
    const int* srcp = eidx;
    const int* dstp = eidx + EE;

    // bump allocator over d_ws
    char* wsp = (char*)d_ws;
    size_t off = 0;
    auto alloc = [&](size_t bytes) -> void* {
        void* p = wsp + off;
        off += (bytes + 255) & ~(size_t)255;
        return p;
    };

    float*  h     = (float*)alloc((size_t)NN * HID * 4);
    float*  hnew  = (float*)alloc((size_t)NN * HID * 4);
    float*  xlb   = (float*)alloc((size_t)NN * HID * 4);
    float*  xrb   = (float*)alloc((size_t)NN * HID * 4);
    float*  eab   = (float*)alloc((size_t)EE * EDIM * 4);
    float*  degb  = (float*)alloc((size_t)NN * 4);
    float*  lea   = (float*)alloc((size_t)NN * EDIM * 4);
    float*  logb  = (float*)alloc((size_t)FF * HEADS * 4);
    float*  mmx   = (float*)alloc((size_t)NN * HEADS * 4);
    float*  den   = (float*)alloc((size_t)NN * HEADS * 4);
    __bf16* h_bf  = (__bf16*)alloc((size_t)NN * HID * 2);
    __bf16* a_bf  = (__bf16*)alloc((size_t)EE * 32 * 2);   // reused: x / padded edge_attr / hg / z
    __bf16* wbf0  = (__bf16*)alloc((size_t)HID * HID * 2);
    __bf16* wbf1  = (__bf16*)alloc((size_t)HID * HID * 2);
    float*  hgsum = (float*)alloc((size_t)NB * HID * 4);
    float*  cntb  = (float*)alloc((size_t)NB * 4);
    float*  hg    = (float*)alloc((size_t)NB * HID * 4);
    float*  zmid  = (float*)alloc((size_t)NB * HID * 4);

    const int T = 256;

    // 1) node embedding: h = x @ node_W + node_b   (WMMA, bf16 inputs, K=64)
    cvt_bf16<<<cdiv((long)NN * NDIM, T), T, 0, stream>>>(x, a_bf, (long)NN * NDIM);
    cvt_bf16<<<cdiv((long)NDIM * HID, T), T, 0, stream>>>(node_W, wbf0, (long)NDIM * HID);
    gemm_bf16_wmma<<<dim3(NN / 16, (HID / 16 + 3) / 4), 128, 0, stream>>>(
        a_bf, wbf0, node_b, h, NN, NDIM, HID);

    // 2) edge embedding: ea = edge_attr @ edge_W + edge_b  (K padded 16 -> 32)
    hipMemsetAsync(a_bf, 0, (size_t)EE * 32 * 2, stream);
    cvt_bf16_pad16to32<<<cdiv((long)EE * EDIM, T), T, 0, stream>>>(eattr, a_bf, EE);
    hipMemsetAsync(wbf0, 0, (size_t)32 * EDIM * 2, stream);
    cvt_bf16<<<1, T, 0, stream>>>(edge_W, wbf0, (long)EDIM * EDIM);
    gemm_bf16_wmma<<<dim3(EE / 16, 1), 128, 0, stream>>>(
        a_bf, wbf0, edge_b, eab, EE, 32, EDIM);

    // 3) self-loop edge attr = mean of incoming (layer-invariant)
    hipMemsetAsync(degb, 0, (size_t)NN * 4, stream);
    hipMemsetAsync(lea,  0, (size_t)NN * EDIM * 4, stream);
    deg_loop_ea<<<cdiv((long)EE * EDIM, T), T, 0, stream>>>(dstp, eab, degb, lea);
    loop_ea_norm<<<cdiv((long)NN * EDIM, T), T, 0, stream>>>(lea, degb);

    // 4) GATv2 layers
    float* hcur = h;
    float* hnxt = hnew;
    for (int l = 0; l < NL; ++l) {
        cvt_bf16<<<cdiv((long)NN * HID, T), T, 0, stream>>>(hcur, h_bf, (long)NN * HID);
        cvt_bf16<<<cdiv((long)HID * HID, T), T, 0, stream>>>(Wl + (long)l * HID * HID, wbf0, (long)HID * HID);
        cvt_bf16<<<cdiv((long)HID * HID, T), T, 0, stream>>>(Wr + (long)l * HID * HID, wbf1, (long)HID * HID);
        gemm_bf16_wmma<<<dim3(NN / 16, (HID / 16 + 3) / 4), 128, 0, stream>>>(
            h_bf, wbf0, bl + l * HID, xlb, NN, HID, HID);
        gemm_bf16_wmma<<<dim3(NN / 16, (HID / 16 + 3) / 4), 128, 0, stream>>>(
            h_bf, wbf1, br + l * HID, xrb, NN, HID, HID);

        hipMemsetAsync(den, 0, (size_t)NN * HEADS * 4, stream);
        fill_f32<<<cdiv((long)NN * HEADS, T), T, 0, stream>>>(mmx, -INFINITY, (long)NN * HEADS);

        attn_logits<<<cdiv((long)FF * HEADS, T), T, 0, stream>>>(
            xlb, xrb, eab, lea, srcp, dstp,
            WeAll + (long)l * EDIM * HID, attAll + (long)l * HEADS * CC, logb, mmx);
        seg_denom<<<cdiv((long)FF * HEADS, T), T, 0, stream>>>(dstp, logb, mmx, den);

        init_hnew<<<cdiv((long)NN * HID, T), T, 0, stream>>>(hcur, cbias + l * HID, hnxt);
        aggregate<<<cdiv((long)FF * CC, T), T, 0, stream>>>(
            xlb, logb, mmx, den, srcp, dstp, hnxt);

        float* tmp = hcur; hcur = hnxt; hnxt = tmp;
    }

    // 5) global mean pool
    hipMemsetAsync(hgsum, 0, (size_t)NB * HID * 4, stream);
    hipMemsetAsync(cntb,  0, (size_t)NB * 4, stream);
    pool_sum<<<cdiv((long)NN * HID, T), T, 0, stream>>>(hcur, batch, hgsum, cntb);
    pool_norm<<<cdiv((long)NB * HID, T), T, 0, stream>>>(hgsum, cntb, hg);

    // 6) output MLP: gelu(hg @ W1 + b1) @ W2 + b2
    cvt_bf16<<<cdiv((long)NB * HID, T), T, 0, stream>>>(hg, a_bf, (long)NB * HID);
    cvt_bf16<<<cdiv((long)HID * HID, T), T, 0, stream>>>(W1, wbf0, (long)HID * HID);
    gemm_bf16_wmma<<<dim3(NB / 16, (HID / 16 + 3) / 4), 128, 0, stream>>>(
        a_bf, wbf0, b1, zmid, NB, HID, HID);
    gelu_exact<<<cdiv((long)NB * HID, T), T, 0, stream>>>(zmid, (long)NB * HID);
    cvt_bf16<<<cdiv((long)NB * HID, T), T, 0, stream>>>(zmid, a_bf, (long)NB * HID);
    cvt_bf16<<<cdiv((long)HID * HID, T), T, 0, stream>>>(W2, wbf1, (long)HID * HID);
    gemm_bf16_wmma<<<dim3(NB / 16, (HID / 16 + 3) / 4), 128, 0, stream>>>(
        a_bf, wbf1, b2, (float*)d_out, NB, HID, HID);
}
